// AutoRegressive_26156350832730
// MI455X (gfx1250) — matrix operand order repaired and checked by reading, not compiled, not run on hardware
//
#include <hip/hip_runtime.h>
#include <hip/hip_bf16.h>

typedef __bf16 bf16_t;
typedef __attribute__((ext_vector_type(16))) __bf16 v16bf;
typedef __attribute__((ext_vector_type(8)))  float  v8f;
typedef __attribute__((ext_vector_type(4)))  unsigned int u32x4;

#define BB   1024          // batch
#define TT   256           // warmup steps
#define II   64            // input signals
#define UU   1024          // units
#define SS   64            // output steps
#define KTOT (II + UU)     // 1088
#define ZN   (4 * UU)      // 4096 gate columns

#define BM      128        // batch rows per block
#define KC      32         // K chunk (one WMMA depth)
#define NCHUNK  (KTOT / KC)       // 34
#define TSTRIDE 40         // staged tile row stride in bf16 (32 data + 8 pad)
#define ZSTRIDE 132        // z tile row stride in f32 (128 + 4 pad)
#define ABUF_BYTES (BM * TSTRIDE * 2)       // 10240
#define SMEM_BYTES (BM * ZSTRIDE * 4)       // 67584 (>= 4*10240 staging)

union Frag { v16bf v; u32x4 q[2]; };

__device__ __forceinline__ v16bf load_frag(const bf16_t* p0, const bf16_t* p1) {
  Frag f;
  f.q[0] = *(const u32x4*)p0;
  f.q[1] = *(const u32x4*)p1;
  return f.v;
}

// Per-lane 16B async copy global->LDS (ASYNCcnt-tracked).
__device__ __forceinline__ void async_b128(unsigned lds_addr, const bf16_t* gsrc) {
  asm volatile("global_load_async_to_lds_b128 %0, %1, off"
               :: "v"(lds_addr), "v"((unsigned long long)(size_t)gsrc)
               : "memory");
}
// Generic pointers to LDS carry the group-relative offset in addr[31:0].
__device__ __forceinline__ unsigned lds_addr_of(const void* p) {
  return (unsigned)(unsigned long long)(size_t)p;
}

__device__ __forceinline__ float sigf(float x) {
  return 1.0f / (1.0f + __expf(-x));
}
__device__ __forceinline__ float tanhfast(float x) {
  x = fminf(fmaxf(x, -15.0f), 15.0f);
  float e = __expf(2.0f * x);
  return (e - 1.0f) / (e + 1.0f);
}

// ---------------------------------------------------------------------------
// One LSTM timestep. Grid: (UU/32, BB/BM) = (32, 8), 256 threads (8 waves).
// Block tile: 128 batch rows x 32 units (128 z cols over 4 gates).
// K loop is double-buffered through LDS via async global->LDS copies.
// ---------------------------------------------------------------------------
__global__ __launch_bounds__(256, 2)
void lstm_step(const bf16_t* __restrict__ Xcur,
               bf16_t* __restrict__ Xnxt,
               float* __restrict__ cstate,
               const bf16_t* __restrict__ Wt,     // [ZN][KTOT] bf16 transposed
               const float* __restrict__ bias,    // [ZN]
               const float* __restrict__ xnext,   // nullptr or f32 rows
               int xstride)
{
  __shared__ __align__(16) unsigned char smem[SMEM_BYTES];

  const int u0   = blockIdx.x * 32;
  const int bm0  = blockIdx.y * BM;
  const int tid  = threadIdx.x;
  const int lane = tid & 31;
  const int wave = tid >> 5;
  const int wm   = wave >> 1;       // 0..3 : 32-row M group (2 x 16-row tiles)
  const int wn   = wave & 1;        // 0..1 : gate pair (i,f) / (g,o)
  const int l15  = lane & 15;
  const int hi   = lane >> 4;

  const unsigned smemBase = lds_addr_of(smem);
  const unsigned aBuf[2] = { smemBase,                  smemBase + ABUF_BYTES };
  const unsigned bBuf[2] = { smemBase + 2 * ABUF_BYTES, smemBase + 3 * ABUF_BYTES };

  // Per-lane fragment metadata (column mapping is gate-major within block).
  int colLoc[4];
  float bv[4];
#pragma unroll
  for (int nt = 0; nt < 4; ++nt) {
    const int gate = wn * 2 + (nt >> 1);
    const int du   = (nt & 1) * 16 + l15;
    colLoc[nt] = gate * 32 + du;                    // 0..127 tile column
    bv[nt]     = bias[gate * UU + u0 + du];
  }

  // Async stage of one K chunk: A tile (128x32) + B tile (128x32), 16B/lane,
  // 2+2 async instructions per thread -> ASYNCcnt += 4 per wave.
  auto stage = [&](int db, int k0) {
#pragma unroll
    for (int i = 0; i < 2; ++i) {
      const int seg = i * 256 + tid;                // 0..511
      const int row = seg >> 2;                     // 0..127
      const int q   = seg & 3;                      // 16B segment in 64B row
      async_b128(aBuf[db] + row * (TSTRIDE * 2) + q * 16,
                 Xcur + (size_t)(bm0 + row) * KTOT + k0 + q * 8);
      const int zc = (row >> 5) * UU + u0 + (row & 31);   // Wt row for col
      async_b128(bBuf[db] + row * (TSTRIDE * 2) + q * 16,
                 Wt + (size_t)zc * KTOT + k0 + q * 8);
    }
  };

  const v8f vzero = {0,0,0,0,0,0,0,0};
  v8f acc[2][4] = {{vzero, vzero, vzero, vzero}, {vzero, vzero, vzero, vzero}};

  stage(0, 0);
  int db = 0;
  for (int chunk = 0; chunk < NCHUNK; ++chunk) {
    if (chunk + 1 < NCHUNK) {
      stage(db ^ 1, (chunk + 1) * KC);
      asm volatile("s_wait_asynccnt 0x4" ::: "memory");  // prev chunk landed
    } else {
      asm volatile("s_wait_asynccnt 0x0" ::: "memory");
    }
    __syncthreads();

    const bf16_t* Ab = (const bf16_t*)(smem + (size_t)db * ABUF_BYTES);
    const bf16_t* Bt = (const bf16_t*)(smem + 2 * ABUF_BYTES + (size_t)db * ABUF_BYTES);

    v16bf afr[2];
#pragma unroll
    for (int mt = 0; mt < 2; ++mt) {
      const bf16_t* ap = Ab + (wm * 32 + mt * 16 + l15) * TSTRIDE + hi * 8;
      afr[mt] = load_frag(ap, ap + 16);             // ds_load_b128 x2
    }
#pragma unroll
    for (int nt = 0; nt < 4; ++nt) {
      const bf16_t* bp = Bt + colLoc[nt] * TSTRIDE + hi * 8;
      v16bf bfr = load_frag(bp, bp + 16);           // ds_load_b128 x2
      acc[0][nt] = __builtin_amdgcn_wmma_f32_16x16x32_bf16(
          false, afr[0], false, bfr, (short)0, acc[0][nt], false, false);
      acc[1][nt] = __builtin_amdgcn_wmma_f32_16x16x32_bf16(
          false, afr[1], false, bfr, (short)0, acc[1][nt], false, false);
    }
    __syncthreads();
    db ^= 1;
  }

  // z tile (+bias) -> LDS (overlays staging buffers; safe after last barrier).
  float* zt = (float*)smem;
#pragma unroll
  for (int mt = 0; mt < 2; ++mt) {
#pragma unroll
    for (int nt = 0; nt < 4; ++nt) {
#pragma unroll
      for (int r = 0; r < 8; ++r) {
        const int rl = wm * 32 + mt * 16 + hi * 8 + r;
        zt[rl * ZSTRIDE + colLoc[nt]] = acc[mt][nt][r] + bv[nt];
      }
    }
  }
  __syncthreads();

  // Gate fusion: 128 rows x 32 units, 16 cells per thread.
  const int uu = tid & 31;
  const int rb = (tid >> 5) * 16;
#pragma unroll
  for (int r = 0; r < 16; ++r) {
    const int rl = rb + r;
    const float zi = zt[rl * ZSTRIDE +       uu];
    const float zf = zt[rl * ZSTRIDE +  32 + uu];
    const float zg = zt[rl * ZSTRIDE +  64 + uu];
    const float zo = zt[rl * ZSTRIDE +  96 + uu];
    const int grow = bm0 + rl;
    const int gu   = u0 + uu;
    const size_t cidx = (size_t)grow * UU + gu;
    const float cc = cstate[cidx];
    const float cn = sigf(zf) * cc + sigf(zi) * tanhfast(zg);
    const float hn = sigf(zo) * tanhfast(cn);
    cstate[cidx] = cn;
    Xnxt[(size_t)grow * KTOT + II + gu] = (bf16_t)hn;
  }

  // Stage next timestep's x columns (warmup only), unit-tile-0 blocks.
  if (xnext != nullptr && blockIdx.x == 0) {
    for (int e = tid; e < BM * II; e += 256) {
      const int r  = e >> 6;
      const int cc = e & (II - 1);
      Xnxt[(size_t)(bm0 + r) * KTOT + cc] =
          (bf16_t)xnext[(size_t)(bm0 + r) * xstride + cc];
    }
  }
}

// ---------------------------------------------------------------------------
// Dense head: p = h @ Wd + bd; writes p (bf16) into Xcur's x-columns and
// gathers output_indices columns to d_out. Grid: BB/64 blocks, 128 threads.
// ---------------------------------------------------------------------------
__global__ __launch_bounds__(128)
void dense_head(bf16_t* __restrict__ Xcur,
                const bf16_t* __restrict__ Wdt,   // [II][UU] bf16 transposed
                const float* __restrict__ bd,     // [II]
                const int* __restrict__ oidx, int nidx,
                float* __restrict__ out, int s)
{
  const int tid  = threadIdx.x;
  const int lane = tid & 31;
  const int wave = tid >> 5;
  const int l15  = lane & 15;
  const int hi   = lane >> 4;
  const int m0   = blockIdx.x * 64 + wave * 16;

  const bf16_t* aBase = Xcur + (size_t)(m0 + l15) * KTOT + II + hi * 8;
  const bf16_t* bBase[4];
#pragma unroll
  for (int nt = 0; nt < 4; ++nt)
    bBase[nt] = Wdt + (size_t)(nt * 16 + l15) * UU + hi * 8;

  const v8f vzero = {0,0,0,0,0,0,0,0};
  v8f acc[4] = {vzero, vzero, vzero, vzero};

  for (int k0 = 0; k0 < UU; k0 += 32) {
    v16bf a = load_frag(aBase + k0, aBase + k0 + 16);
#pragma unroll
    for (int nt = 0; nt < 4; ++nt) {
      v16bf bfr = load_frag(bBase[nt] + k0, bBase[nt] + k0 + 16);
      acc[nt] = __builtin_amdgcn_wmma_f32_16x16x32_bf16(
          false, a, false, bfr, (short)0, acc[nt], false, false);
    }
  }

#pragma unroll
  for (int nt = 0; nt < 4; ++nt) {
    const int col = nt * 16 + l15;
    const float bb = bd[col];
#pragma unroll
    for (int r = 0; r < 8; ++r) {
      const int row = m0 + hi * 8 + r;
      const float v = acc[nt][r] + bb;
      Xcur[(size_t)row * KTOT + col] = (bf16_t)v;       // feedback input
      for (int j = 0; j < nidx; ++j)
        if (col == oidx[j])
          out[(size_t)row * (SS * nidx) + (size_t)s * nidx + j] = v;
    }
  }
}

// ---------------------------------------------------------------------------
// Prep kernels: bf16 weight packing (transposed) + state init.
// ---------------------------------------------------------------------------
__global__ void prep_weights(const float* __restrict__ Wk,
                             const float* __restrict__ Wr,
                             bf16_t* __restrict__ Wt)
{
  const int total = ZN * KTOT;
  for (int e = blockIdx.x * blockDim.x + threadIdx.x; e < total;
       e += gridDim.x * blockDim.x) {
    const int col = e / KTOT;
    const int k   = e - col * KTOT;
    const float v = (k < II) ? Wk[(size_t)k * ZN + col]
                             : Wr[(size_t)(k - II) * ZN + col];
    Wt[e] = (bf16_t)v;
  }
}

__global__ void prep_wd(const float* __restrict__ Wd, bf16_t* __restrict__ Wdt)
{
  const int total = II * UU;
  for (int e = blockIdx.x * blockDim.x + threadIdx.x; e < total;
       e += gridDim.x * blockDim.x) {
    const int n = e / UU;
    const int k = e - n * UU;
    Wdt[e] = (bf16_t)Wd[(size_t)k * II + n];
  }
}

__global__ void prep_state(const float* __restrict__ inputs,
                           bf16_t* __restrict__ X0,
                           float* __restrict__ cstate)
{
  const int tX = BB * KTOT;
  for (int e = blockIdx.x * blockDim.x + threadIdx.x; e < tX;
       e += gridDim.x * blockDim.x) {
    const int row = e / KTOT;
    const int k   = e - row * KTOT;
    X0[e] = (k < II) ? (bf16_t)inputs[(size_t)row * (TT * II) + k]
                     : (bf16_t)0.0f;
  }
  const int tC = BB * UU;
  for (int e = blockIdx.x * blockDim.x + threadIdx.x; e < tC;
       e += gridDim.x * blockDim.x)
    cstate[e] = 0.0f;
}

// ---------------------------------------------------------------------------
extern "C" void kernel_launch(void* const* d_in, const int* in_sizes, int n_in,
                              void* d_out, int out_size, void* d_ws,
                              size_t ws_size, hipStream_t stream)
{
  (void)n_in; (void)out_size; (void)ws_size;
  const float* inputs = (const float*)d_in[0];
  const float* Wk     = (const float*)d_in[1];
  const float* Wr     = (const float*)d_in[2];
  const float* bz     = (const float*)d_in[3];
  const float* Wd     = (const float*)d_in[4];
  const float* bd     = (const float*)d_in[5];
  const int*   oidx   = (const int*)d_in[6];
  const int    nidx   = in_sizes[6];
  float* out = (float*)d_out;

  char* base = (char*)d_ws;
  size_t off = 0;
  auto carve = [&](size_t bytes) -> void* {
    void* p = base + off;
    off += (bytes + 255) & ~(size_t)255;
    return p;
  };
  bf16_t* Wt  = (bf16_t*)carve((size_t)ZN * KTOT * sizeof(bf16_t)); // 8.9 MB
  bf16_t* Wdt = (bf16_t*)carve((size_t)II * UU * sizeof(bf16_t));   // 128 KB
  bf16_t* X0  = (bf16_t*)carve((size_t)BB * KTOT * sizeof(bf16_t)); // 2.2 MB
  bf16_t* X1  = (bf16_t*)carve((size_t)BB * KTOT * sizeof(bf16_t)); // 2.2 MB
  float*  cst = (float*)carve((size_t)BB * UU * sizeof(float));     // 4.2 MB

  prep_weights<<<2048, 256, 0, stream>>>(Wk, Wr, Wt);
  prep_wd<<<256, 256, 0, stream>>>(Wd, Wdt);
  prep_state<<<1024, 256, 0, stream>>>(inputs, X0, cst);

  bf16_t* Xc = X0;
  bf16_t* Xn = X1;
  const dim3 sgrid(UU / 32, BB / BM);   // (32, 8) = 256 blocks/step

  for (int t = 0; t < TT; ++t) {
    const float* xnext = (t + 1 < TT) ? (inputs + (size_t)(t + 1) * II)
                                      : nullptr;
    lstm_step<<<sgrid, 256, 0, stream>>>(Xc, Xn, cst, Wt, bz, xnext, TT * II);
    bf16_t* tmp = Xc; Xc = Xn; Xn = tmp;
  }

  dense_head<<<BB / 64, 128, 0, stream>>>(Xc, Wdt, bd, oidx, nidx, out, 0);
  for (int s = 1; s < SS; ++s) {
    lstm_step<<<sgrid, 256, 0, stream>>>(Xc, Xn, cst, Wt, bz, nullptr, 0);
    bf16_t* tmp = Xc; Xc = Xn; Xn = tmp;
    dense_head<<<BB / 64, 128, 0, stream>>>(Xc, Wdt, bd, oidx, nidx, out, s);
  }
}
